// FruityGNNLayer_69312182222993
// MI455X (gfx1250) — matrix operand order repaired and checked by reading, not compiled, not run on hardware
//
#include <hip/hip_runtime.h>
#include <stdint.h>

// ---------------- problem constants ----------------
#define NNODES 50000
#define NEDGES 300000
#define NGRAPH 64
#define ND 256
#define ED 256
#define GD 128

// ---------------- types ----------------
typedef __attribute__((ext_vector_type(16))) __bf16    v16bf;
typedef __attribute__((ext_vector_type(8)))  float     v8f;
typedef __attribute__((ext_vector_type(4)))  uint32_t  v4u;

union FragU { v16bf v; v4u q[2]; uint32_t u[8]; };

// ---------------- helpers ----------------
__device__ __forceinline__ uint16_t f2bf(float x) {
    uint32_t u = __float_as_uint(x);
    return (uint16_t)((u + 0x7FFFu + ((u >> 16) & 1u)) >> 16);
}
__device__ __forceinline__ uint32_t pack2bf(float a, float b) {
    return (uint32_t)f2bf(a) | ((uint32_t)f2bf(b) << 16);
}

// A-fragment swizzle: 16-bit A 16x32 layout (ISA 7.12.2).
// lane(0..15)=row M, halves: lanes0-15 K=0..7,16..23 ; lanes16-31 K=8..15,24..31
// Storage: dword index = ((kc*32 + h*16 + m) * 8) + v  (v = which dword pair)
__device__ __forceinline__ int a_swz(int m, int p /* = k/2 */) {
    int k   = p << 1;
    int kc  = k >> 5;
    int kp  = k & 31;
    int hi  = kp >> 4;            // 1 -> v in 4..7
    int rem = kp & 15;
    int h   = rem >> 3;           // lane half
    int v   = hi * 4 + ((rem & 7) >> 1);
    return (((kc << 5) + (h << 4) + m) << 3) + v;
}

__device__ __forceinline__ v16bf load_afrag(const uint32_t* lds, int kc) {
    int lane = threadIdx.x & 31;
    const v4u* p = (const v4u*)(lds + (((kc << 5) + lane) << 3));
    FragU f; f.q[0] = p[0]; f.q[1] = p[1];
    return f.v;
}

__device__ __forceinline__ v16bf load_bfrag(const uint32_t* wpk, int tile) {
    int lane = threadIdx.x & 31;
    const v4u* p = (const v4u*)(wpk + (((size_t)tile << 5) + (size_t)lane) * 8);
    FragU f; f.q[0] = p[0]; f.q[1] = p[1];
    return f.v;
}

__device__ __forceinline__ v8f wmma_bf16(v16bf a, v16bf b, v8f c) {
    return __builtin_amdgcn_wmma_f32_16x16x32_bf16(false, a, false, b, (short)0, c, false, false);
}

// Store one D element into an A-swizzled LDS h-tile (element becomes K of next layer)
__device__ __forceinline__ void store_h(uint16_t* ldsH, int m, int k, float val) {
    int di = a_swz(m, k >> 1);
    ldsH[di * 2 + (k & 1)] = f2bf(val);
}

__device__ __forceinline__ void relu_store_h(uint16_t* ldsH, v8f acc, int ntile) {
    int lane = threadIdx.x & 31;
    int n  = ntile * 16 + (lane & 15);
    int mb = (lane >> 4) * 8;
    #pragma unroll
    for (int j = 0; j < 8; ++j) store_h(ldsH, mb + j, n, fmaxf(acc[j], 0.0f));
}

// Two N-tiles per wave (N = 256, 8 waves). D = A*W^T + bias
template<int K>
__device__ __forceinline__ void mlp2(const uint32_t* ldsA, const uint32_t* wpk,
                                     const float* __restrict__ bias, v8f& o0, v8f& o1) {
    constexpr int KC = K >> 5;
    int wv = threadIdx.x >> 5;
    v8f a0 = {0.f,0.f,0.f,0.f,0.f,0.f,0.f,0.f};
    v8f a1 = a0;
    #pragma unroll
    for (int kc = 0; kc < KC; ++kc) {
        v16bf a  = load_afrag(ldsA, kc);
        v16bf b0 = load_bfrag(wpk,  wv      * KC + kc);
        v16bf b1 = load_bfrag(wpk, (wv + 8) * KC + kc);
        a0 = wmma_bf16(a, b0, a0);
        a1 = wmma_bf16(a, b1, a1);
    }
    int nl = threadIdx.x & 15;
    float bb0 = bias[wv * 16 + nl];
    float bb1 = bias[(wv + 8) * 16 + nl];
    #pragma unroll
    for (int j = 0; j < 8; ++j) { a0[j] += bb0; a1[j] += bb1; }
    o0 = a0; o1 = a1;
}

// One N-tile per wave (N = 128, 8 waves)
template<int K>
__device__ __forceinline__ v8f mlp1(const uint32_t* ldsA, const uint32_t* wpk,
                                    const float* __restrict__ bias) {
    constexpr int KC = K >> 5;
    int wv = threadIdx.x >> 5;
    v8f acc = {0.f,0.f,0.f,0.f,0.f,0.f,0.f,0.f};
    #pragma unroll
    for (int kc = 0; kc < KC; ++kc) {
        v16bf a = load_afrag(ldsA, kc);
        v16bf b = load_bfrag(wpk, wv * KC + kc);
        acc = wmma_bf16(a, b, acc);
    }
    float bb = bias[wv * 16 + (threadIdx.x & 15)];
    #pragma unroll
    for (int j = 0; j < 8; ++j) acc[j] += bb;
    return acc;
}

// Scatter-add 8 D-elements (same column n, rows mb..mb+7) with run-merging
// over equal group ids (batch is sorted for nodes -> usually a single atomic).
__device__ __forceinline__ void scatter8(float* __restrict__ agg, const int* gid,
                                         int mb, v8f v, int n, int cols) {
    int cur = gid[mb];
    float s = v[0];
    #pragma unroll
    for (int j = 1; j < 8; ++j) {
        int g = gid[mb + j];
        if (g == cur) { s += v[j]; }
        else {
            atomicAdd(&agg[(size_t)cur * cols + n], s);
            cur = g; s = v[j];
        }
    }
    atomicAdd(&agg[(size_t)cur * cols + n], s);
}

// ---------------- prep kernels ----------------
__global__ void zero_kernel(float* __restrict__ p, int n) {
    int i = blockIdx.x * blockDim.x + threadIdx.x;
    int s = gridDim.x * blockDim.x;
    for (; i < n; i += s) p[i] = 0.0f;
}

__global__ void cvt2_kernel(const float* __restrict__ src, uint32_t* __restrict__ dst, int npairs) {
    int i = blockIdx.x * blockDim.x + threadIdx.x;
    int s = gridDim.x * blockDim.x;
    for (; i < npairs; i += s) dst[i] = pack2bf(src[2 * i], src[2 * i + 1]);
}

// Pack torch-style weight w[N][K] (f32) into per-lane WMMA B-fragment order (bf16).
// B 16-bit 32x16 layout: lane = h*16 + n ; VGPR v holds K = kc*32 + (v/4)*16 + h*8 + 2*(v%4) (,+1)
__global__ void pack_w_kernel(const float* __restrict__ w, uint32_t* __restrict__ dst,
                              int N, int K) {
    int total = (N * K) >> 1;
    int KC = K >> 5;
    int i = blockIdx.x * blockDim.x + threadIdx.x;
    int s = gridDim.x * blockDim.x;
    for (; i < total; i += s) {
        int v    = i & 7;
        int lane = (i >> 3) & 31;
        int tile = i >> 8;
        int kc   = tile % KC;
        int nt   = tile / KC;
        int n    = nt * 16 + (lane & 15);
        int k    = (kc << 5) + ((v >> 2) << 4) + ((lane >> 4) << 3) + ((v & 3) << 1);
        const float* row = w + (size_t)n * K + k;
        dst[i] = pack2bf(row[0], row[1]);
    }
}

// ---------------- message MLP + scatter into agg_msg ----------------
__global__ __launch_bounds__(256) void msg_kernel(
    const uint16_t* __restrict__ x_bf, const uint16_t* __restrict__ ea_bf,
    const int* __restrict__ ei,
    const uint32_t* __restrict__ w1pk, const float* __restrict__ b1,
    const uint32_t* __restrict__ w2pk, const float* __restrict__ b2,
    float* __restrict__ agg_msg)
{
    __shared__ __attribute__((aligned(16))) uint32_t ldsA[4096];   // 16 x 512 bf16
    __shared__ __attribute__((aligned(16))) uint32_t ldsH32[2048]; // 16 x 256 bf16
    __shared__ int tgt[16];
    uint16_t* ldsH = (uint16_t*)ldsH32;

    int t  = threadIdx.x;
    int e0 = blockIdx.x * 16;
    int m  = t >> 4, sub = t & 15;
    int e  = e0 + m;
    int src = ei[e];
    const uint32_t* xr = (const uint32_t*)(x_bf + (size_t)src * ND);
    const uint32_t* er = (const uint32_t*)(ea_bf + (size_t)e * ED);
    #pragma unroll
    for (int j = 0; j < 16; ++j) {
        int p = sub + j * 16;                 // K = 512 -> 256 pairs
        uint32_t d = (p < 128) ? xr[p] : er[p - 128];
        ldsA[a_swz(m, p)] = d;
    }
    if (t < 16) tgt[t] = ei[NEDGES + e0 + t];
    __syncthreads();

    v8f h0, h1;
    mlp2<512>(ldsA, w1pk, b1, h0, h1);
    relu_store_h(ldsH, h0, t >> 5);
    relu_store_h(ldsH, h1, (t >> 5) + 8);
    __syncthreads();

    v8f o0, o1;
    mlp2<256>((const uint32_t*)ldsH32, w2pk, b2, o0, o1);

    int lane = t & 31;
    int mb = (lane >> 4) * 8;
    int n0 = (t >> 5) * 16 + (lane & 15);
    scatter8(agg_msg, tgt, mb, o0, n0, ND);
    scatter8(agg_msg, tgt, mb, o1, n0 + 128, ND);
}

// ---------------- node MLP -> x_updated + agg_nodes ----------------
__global__ __launch_bounds__(256) void node_kernel(
    const uint16_t* __restrict__ x_bf, const float* __restrict__ agg_msg,
    const uint16_t* __restrict__ gcv_bf, const int* __restrict__ batch,
    const uint32_t* __restrict__ w1pk, const float* __restrict__ b1,
    const uint32_t* __restrict__ w2pk, const float* __restrict__ b2,
    float* __restrict__ out_x, float* __restrict__ agg_nodes)
{
    __shared__ __attribute__((aligned(16))) uint32_t ldsA[5120];   // 16 x 640 bf16
    __shared__ __attribute__((aligned(16))) uint32_t ldsH32[2048];
    __shared__ int gb[16];
    uint16_t* ldsH = (uint16_t*)ldsH32;

    int t  = threadIdx.x;
    int n0 = blockIdx.x * 16;
    int m  = t >> 4, sub = t & 15;
    int id = n0 + m;
    int bg = batch[id];
    if (sub == 0) gb[m] = bg;
    const uint32_t* xr = (const uint32_t*)(x_bf + (size_t)id * ND);
    const float*    ar = agg_msg + (size_t)id * ND;
    const uint32_t* gr = (const uint32_t*)(gcv_bf + (size_t)bg * GD);
    #pragma unroll
    for (int j = 0; j < 20; ++j) {
        int p = sub + j * 16;                 // K = 640 -> 320 pairs
        uint32_t d;
        if (p < 128)      d = xr[p];
        else if (p < 256) { int q = (p - 128) * 2; d = pack2bf(ar[q], ar[q + 1]); }
        else              d = gr[p - 256];
        ldsA[a_swz(m, p)] = d;
    }
    __syncthreads();

    v8f h0, h1;
    mlp2<640>(ldsA, w1pk, b1, h0, h1);
    relu_store_h(ldsH, h0, t >> 5);
    relu_store_h(ldsH, h1, (t >> 5) + 8);
    __syncthreads();

    v8f o0, o1;
    mlp2<256>((const uint32_t*)ldsH32, w2pk, b2, o0, o1);

    int lane = t & 31;
    int mb = (lane >> 4) * 8;
    int nn = (t >> 5) * 16 + (lane & 15);
    #pragma unroll
    for (int j = 0; j < 8; ++j) {
        int mm = mb + j;
        out_x[(size_t)(n0 + mm) * ND + nn]       = o0[j];
        out_x[(size_t)(n0 + mm) * ND + nn + 128] = o1[j];
    }
    scatter8(agg_nodes, gb, mb, o0, nn, ND);
    scatter8(agg_nodes, gb, mb, o1, nn + 128, ND);
}

// ---------------- edge MLP -> edge_updated + agg_edges ----------------
__global__ __launch_bounds__(256) void edge_kernel(
    const uint16_t* __restrict__ x_bf, const uint16_t* __restrict__ ea_bf,
    const uint16_t* __restrict__ gcv_bf,
    const int* __restrict__ ei, const int* __restrict__ batch,
    const uint32_t* __restrict__ w1pk, const float* __restrict__ b1,
    const uint32_t* __restrict__ w2pk, const float* __restrict__ b2,
    float* __restrict__ out_e, float* __restrict__ agg_edges)
{
    __shared__ __attribute__((aligned(16))) uint32_t ldsA[7168];   // 16 x 896 bf16
    __shared__ __attribute__((aligned(16))) uint32_t ldsH32[2048];
    __shared__ int gb[16];
    uint16_t* ldsH = (uint16_t*)ldsH32;

    int t  = threadIdx.x;
    int e0 = blockIdx.x * 16;
    int m  = t >> 4, sub = t & 15;
    int e  = e0 + m;
    int src = ei[e];
    int dst = ei[NEDGES + e];
    int bg  = batch[src];
    if (sub == 0) gb[m] = bg;
    const uint32_t* er = (const uint32_t*)(ea_bf + (size_t)e * ED);
    const uint32_t* xs = (const uint32_t*)(x_bf + (size_t)src * ND);
    const uint32_t* xt = (const uint32_t*)(x_bf + (size_t)dst * ND);
    const uint32_t* gr = (const uint32_t*)(gcv_bf + (size_t)bg * GD);
    #pragma unroll
    for (int j = 0; j < 28; ++j) {
        int p = sub + j * 16;                 // K = 896 -> 448 pairs
        uint32_t d;
        if (p < 128)      d = er[p];
        else if (p < 256) d = xs[p - 128];
        else if (p < 384) d = xt[p - 256];
        else              d = gr[p - 384];
        ldsA[a_swz(m, p)] = d;
    }
    __syncthreads();

    v8f h0, h1;
    mlp2<896>(ldsA, w1pk, b1, h0, h1);
    relu_store_h(ldsH, h0, t >> 5);
    relu_store_h(ldsH, h1, (t >> 5) + 8);
    __syncthreads();

    v8f o0, o1;
    mlp2<256>((const uint32_t*)ldsH32, w2pk, b2, o0, o1);

    int lane = t & 31;
    int mb = (lane >> 4) * 8;
    int nn = (t >> 5) * 16 + (lane & 15);
    #pragma unroll
    for (int j = 0; j < 8; ++j) {
        int mm = mb + j;
        out_e[(size_t)(e0 + mm) * ED + nn]       = o0[j];
        out_e[(size_t)(e0 + mm) * ED + nn + 128] = o1[j];
    }
    scatter8(agg_edges, gb, mb, o0, nn, ED);
    scatter8(agg_edges, gb, mb, o1, nn + 128, ED);
}

// ---------------- global MLP (4 blocks of 16 graphs) ----------------
__global__ __launch_bounds__(256) void glob_kernel(
    const uint16_t* __restrict__ gcv_bf,
    const float* __restrict__ agg_nodes, const float* __restrict__ agg_edges,
    const uint32_t* __restrict__ w1pk, const float* __restrict__ b1,
    const uint32_t* __restrict__ w2pk, const float* __restrict__ b2,
    float* __restrict__ out_g)
{
    __shared__ __attribute__((aligned(16))) uint32_t ldsA[5120];   // 16 x 640 bf16
    __shared__ __attribute__((aligned(16))) uint32_t ldsH32[1024]; // 16 x 128 bf16
    uint16_t* ldsH = (uint16_t*)ldsH32;

    int t  = threadIdx.x;
    int g0 = blockIdx.x * 16;
    int m  = t >> 4, sub = t & 15;
    int g  = g0 + m;
    const uint32_t* gr = (const uint32_t*)(gcv_bf + (size_t)g * GD);
    const float*    an = agg_nodes + (size_t)g * ND;
    const float*    ae = agg_edges + (size_t)g * ED;
    #pragma unroll
    for (int j = 0; j < 20; ++j) {
        int p = sub + j * 16;                 // K = 640 -> 320 pairs
        uint32_t d;
        if (p < 64)       d = gr[p];
        else if (p < 192) { int q = (p - 64) * 2;  d = pack2bf(an[q], an[q + 1]); }
        else              { int q = (p - 192) * 2; d = pack2bf(ae[q], ae[q + 1]); }
        ldsA[a_swz(m, p)] = d;
    }
    __syncthreads();

    v8f h = mlp1<640>(ldsA, w1pk, b1);
    relu_store_h(ldsH, h, t >> 5);
    __syncthreads();

    v8f o = mlp1<128>((const uint32_t*)ldsH32, w2pk, b2);

    int lane = t & 31;
    int mb = (lane >> 4) * 8;
    int n  = (t >> 5) * 16 + (lane & 15);
    #pragma unroll
    for (int j = 0; j < 8; ++j)
        out_g[(size_t)(g0 + mb + j) * GD + n] = o[j];
}

// ---------------- launcher ----------------
extern "C" void kernel_launch(void* const* d_in, const int* in_sizes, int n_in,
                              void* d_out, int out_size, void* d_ws, size_t ws_size,
                              hipStream_t stream) {
    (void)in_sizes; (void)n_in; (void)out_size; (void)ws_size;

    const float* x       = (const float*)d_in[0];
    const int*   ei      = (const int*)  d_in[1];
    const float* ea      = (const float*)d_in[2];
    const float* gcv     = (const float*)d_in[3];
    const int*   batch   = (const int*)  d_in[4];
    const float* msg_w1  = (const float*)d_in[5];
    const float* msg_b1  = (const float*)d_in[6];
    const float* msg_w2  = (const float*)d_in[7];
    const float* msg_b2  = (const float*)d_in[8];
    const float* node_w1 = (const float*)d_in[9];
    const float* node_b1 = (const float*)d_in[10];
    const float* node_w2 = (const float*)d_in[11];
    const float* node_b2 = (const float*)d_in[12];
    const float* edge_w1 = (const float*)d_in[13];
    const float* edge_b1 = (const float*)d_in[14];
    const float* edge_w2 = (const float*)d_in[15];
    const float* edge_b2 = (const float*)d_in[16];
    const float* glob_w1 = (const float*)d_in[17];
    const float* glob_b1 = (const float*)d_in[18];
    const float* glob_w2 = (const float*)d_in[19];
    const float* glob_b2 = (const float*)d_in[20];

    float* out_x = (float*)d_out;
    float* out_e = out_x + (size_t)NNODES * ND;
    float* out_g = out_e + (size_t)NEDGES * ED;

    // workspace layout
    char* ws = (char*)d_ws;
    size_t off = 0;
    float* agg_msg   = (float*)(ws + off); off += (size_t)NNODES * ND * 4;
    float* agg_nodes = (float*)(ws + off); off += (size_t)NGRAPH * ND * 4;
    float* agg_edges = (float*)(ws + off); off += (size_t)NGRAPH * ED * 4;
    uint16_t* x_bf   = (uint16_t*)(ws + off); off += (size_t)NNODES * ND * 2;
    uint16_t* ea_bf  = (uint16_t*)(ws + off); off += (size_t)NEDGES * ED * 2;
    uint16_t* gcv_bf = (uint16_t*)(ws + off); off += (size_t)NGRAPH * GD * 2;
    uint32_t* pk_msg1  = (uint32_t*)(ws + off); off += (size_t)(256 * 512 / 2) * 4;
    uint32_t* pk_msg2  = (uint32_t*)(ws + off); off += (size_t)(256 * 256 / 2) * 4;
    uint32_t* pk_node1 = (uint32_t*)(ws + off); off += (size_t)(256 * 640 / 2) * 4;
    uint32_t* pk_node2 = (uint32_t*)(ws + off); off += (size_t)(256 * 256 / 2) * 4;
    uint32_t* pk_edge1 = (uint32_t*)(ws + off); off += (size_t)(256 * 896 / 2) * 4;
    uint32_t* pk_edge2 = (uint32_t*)(ws + off); off += (size_t)(256 * 256 / 2) * 4;
    uint32_t* pk_glob1 = (uint32_t*)(ws + off); off += (size_t)(128 * 640 / 2) * 4;
    uint32_t* pk_glob2 = (uint32_t*)(ws + off); off += (size_t)(128 * 128 / 2) * 4;

    const int B = 256;

    // zero all accumulators (contiguous region)
    zero_kernel<<<2048, B, 0, stream>>>(agg_msg, NNODES * ND + 2 * NGRAPH * ND);

    // bf16 conversions
    cvt2_kernel<<<2048, B, 0, stream>>>(x,   (uint32_t*)x_bf,   NNODES * ND / 2);
    cvt2_kernel<<<4096, B, 0, stream>>>(ea,  (uint32_t*)ea_bf,  NEDGES * ED / 2);
    cvt2_kernel<<<16,   B, 0, stream>>>(gcv, (uint32_t*)gcv_bf, NGRAPH * GD / 2);

    // weight packing into WMMA B-fragment order
    pack_w_kernel<<<256, B, 0, stream>>>(msg_w1,  pk_msg1,  256, 512);
    pack_w_kernel<<<128, B, 0, stream>>>(msg_w2,  pk_msg2,  256, 256);
    pack_w_kernel<<<320, B, 0, stream>>>(node_w1, pk_node1, 256, 640);
    pack_w_kernel<<<128, B, 0, stream>>>(node_w2, pk_node2, 256, 256);
    pack_w_kernel<<<448, B, 0, stream>>>(edge_w1, pk_edge1, 256, 896);
    pack_w_kernel<<<128, B, 0, stream>>>(edge_w2, pk_edge2, 256, 256);
    pack_w_kernel<<<160, B, 0, stream>>>(glob_w1, pk_glob1, 128, 640);
    pack_w_kernel<<<32,  B, 0, stream>>>(glob_w2, pk_glob2, 128, 128);

    // main pipeline
    msg_kernel <<<NEDGES / 16, B, 0, stream>>>(x_bf, ea_bf, ei,
                                               pk_msg1, msg_b1, pk_msg2, msg_b2, agg_msg);
    node_kernel<<<NNODES / 16, B, 0, stream>>>(x_bf, agg_msg, gcv_bf, batch,
                                               pk_node1, node_b1, pk_node2, node_b2,
                                               out_x, agg_nodes);
    edge_kernel<<<NEDGES / 16, B, 0, stream>>>(x_bf, ea_bf, gcv_bf, ei, batch,
                                               pk_edge1, edge_b1, pk_edge2, edge_b2,
                                               out_e, agg_edges);
    glob_kernel<<<NGRAPH / 16, B, 0, stream>>>(gcv_bf, agg_nodes, agg_edges,
                                               pk_glob1, glob_b1, pk_glob2, glob_b2, out_g);
}
